// MetaUpsampler_33483565039833
// MI455X (gfx1250) — matrix operand order repaired and checked by reading, not compiled, not run on hardware
//
#include <hip/hip_runtime.h>
#include <math.h>

typedef _Float16 half_t;
typedef __attribute__((ext_vector_type(16))) _Float16 v16h;
typedef __attribute__((ext_vector_type(8)))  _Float16 v8h;
typedef __attribute__((ext_vector_type(8)))  float    v8f;

#define Bn 2
#define Cc 64
#define Hh 128
#define Ww 128
#define OH 256
#define OW 256
#define HW (Hh*Ww)

// workspace layout (units: halfs)
#define WK1_OFF 0            // 20 ktiles * 8 ntiles * 32 lanes * 16 halfs = 81920
#define WK2_OFF 81920        // 4 * 512 = 2048
#define WR1_OFF 83968        // 2*4*512 = 4096
#define WR2_OFF 88064        // 2*512   = 1024
#define MEMB_OFF 89088       // 4*64    = 256
#define WS_HALFS 89344

__device__ __forceinline__ float gelu_f(float x) {
    return 0.5f * x * (1.0f + erff(x * 0.70710678118654752f));
}

// ---------------------------------------------------------------------------
// Meta MLP: only 4 distinct (du,dv) parities exist for scale=2.
// ---------------------------------------------------------------------------
__global__ void prep_meta(const float* __restrict__ w1m, const float* __restrict__ b1m,
                          const float* __restrict__ w2m, const float* __restrict__ b2m,
                          half_t* __restrict__ wsH) {
    __shared__ float hbuf[4 * 64];
    int t = threadIdx.x;
    int pidx = t >> 6, o = t & 63;
    float du = (pidx & 1) ? -0.25f : 0.25f;
    float dv = (pidx & 2) ? -0.25f : 0.25f;
    float m[5] = {2.0f, du, dv, 0.5f, 0.3f};
    float enc[85];
    const float PI = 3.14159265358979323846f;
    for (int i = 0; i < 5; i++) {
        enc[i * 17 + 0] = m[i];
        float f = PI;
        for (int j = 0; j < 8; j++) {
            float xb = m[i] * f;
            enc[i * 17 + 1 + j] = sinf(xb);
            enc[i * 17 + 9 + j] = cosf(xb);
            f *= 2.0f;
        }
    }
    float acc = b1m[o];
    for (int k = 0; k < 85; k++) acc += w1m[o * 85 + k] * enc[k];
    hbuf[t] = gelu_f(acc);
    __syncthreads();
    float out = b2m[o];
    for (int k = 0; k < 64; k++) out += w2m[o * 64 + k] * hbuf[pidx * 64 + k];
    wsH[MEMB_OFF + t] = (half_t)out;
}

// ---------------------------------------------------------------------------
// Pack fp32 weights into f16 WMMA B-fragments.
// B 32x16 layout: lanes 0-15 hold K0..7,K16..23; lanes 16-31 hold K8..15,K24..31.
// wk1 K-dim is PERMUTED to tap-major: k' = tap*64 + ch  (tap=k%9, ch=k/9)
// so the main kernel's A gather needs no /9. k' >= 576 (m_emb) unchanged.
// ---------------------------------------------------------------------------
__global__ void prep_pack(const float* __restrict__ wk1, const float* __restrict__ wk2,
                          const float* __restrict__ wr1, const float* __restrict__ wr2,
                          half_t* __restrict__ wsH) {
    int tid = blockIdx.x * blockDim.x + threadIdx.x;
    if (tid < 81920) {
        int e = tid;
        int kt = e >> 12;
        int r  = e & 4095;
        int nt = r >> 9;
        int r2 = r & 511;
        int lane = r2 >> 4, h = r2 & 15;
        int kloc = ((h & 8) << 1) + ((lane >> 4) << 3) + (h & 7);
        int n = lane & 15;
        int kp = kt * 32 + kloc;                       // permuted K index
        int korig = (kp < 576) ? ((kp & 63) * 9 + (kp >> 6)) : kp;
        wsH[WK1_OFF + tid] = (half_t)wk1[(nt * 16 + n) * 640 + korig];
    } else if (tid < 81920 + 2048) {
        int e = tid - 81920;
        int kt = e >> 9, r2 = e & 511;
        int lane = r2 >> 4, h = r2 & 15;
        int kloc = ((h & 8) << 1) + ((lane >> 4) << 3) + (h & 7);
        int n = lane & 15;
        float v = (n < 9) ? wk2[n * 128 + kt * 32 + kloc] : 0.0f;
        wsH[WK2_OFF + e] = (half_t)v;
    } else if (tid < 81920 + 2048 + 4096) {
        int e = tid - (81920 + 2048);
        int kt = e >> 11;
        int r  = e & 2047;
        int nt = r >> 9, r2 = r & 511;
        int lane = r2 >> 4, h = r2 & 15;
        int kloc = ((h & 8) << 1) + ((lane >> 4) << 3) + (h & 7);
        int n = lane & 15;
        wsH[WR1_OFF + e] = (half_t)wr1[(nt * 16 + n) * 64 + kt * 32 + kloc];
    } else if (tid < 81920 + 2048 + 4096 + 1024) {
        int e = tid - (81920 + 2048 + 4096);
        int kt = e >> 9, r2 = e & 511;
        int lane = r2 >> 4, h = r2 & 15;
        int kloc = ((h & 8) << 1) + ((lane >> 4) << 3) + (h & 7);
        int n = lane & 15;
        float v = (n < 3) ? wr2[n * 64 + kt * 32 + kloc] : 0.0f;
        wsH[WR2_OFF + e] = (half_t)v;
    }
}

// ---------------------------------------------------------------------------
// per-wave LDS region layout (bytes):
//   cache : [3 rows][16 cols][64 ch] f16  -> 6144
//   hid   : 16q x 128 f16                 -> 4096  (reused as hid2 in stage 3/4)
//   w2b   : 16x16 f32 raw kernel weights  -> 1024
//   wbuf  : 16x12 f32 softmaxed weights   ->  768
//   fqb   : 16q x 64 f16 f_q              -> 2048
#define CACHE_B 0
#define HID_B   6144
#define W2B_B   10240
#define WBUF_B  11264
#define FQB_B   12032
#define WAVE_LDS 14336
#define WAVES 4

// A-fragment builder for stage 1 (kt is a compile-time constant after unroll).
// 16-bit A layout: lanes0-15 (M=lane) halfs 0..7 = K0..7, halfs 8..15 = K16..23;
// lanes16-31 hold K8..15 / K24..31.
__device__ __forceinline__ v16h buildA(int kt, const half_t* cacheH,
                                       const half_t* memb, int cxloc, int hi) {
    if (kt < 18) {
        int t  = kt >> 1;          // tap index (constant per unrolled kt)
        int ky = t / 3, kx = t - ky * 3;
        int cbase = ((kt & 1) << 5) + hi * 8;
        const v8h* cp = (const v8h*)(cacheH + ((ky * 16 + cxloc + kx) << 6) + cbase);
        v8h lo = cp[0], hh = cp[2];
        return __builtin_shufflevector(lo, hh, 0,1,2,3,4,5,6,7,8,9,10,11,12,13,14,15);
    } else {
        const v8h* mb = (const v8h*)memb;
        int i0 = (kt - 18) * 4 + hi;
        v8h lo = mb[i0], hh = mb[i0 + 2];
        return __builtin_shufflevector(lo, hh, 0,1,2,3,4,5,6,7,8,9,10,11,12,13,14,15);
    }
}

__global__ __launch_bounds__(128) void main_kernel(
        const float* __restrict__ feat,
        const float* __restrict__ bk1, const float* __restrict__ bk2,
        const float* __restrict__ br1, const float* __restrict__ br2,
        const half_t* __restrict__ wsH, float* __restrict__ out) {
    __shared__ __align__(16) unsigned char ldsRaw[WAVES * WAVE_LDS];
    int lane = threadIdx.x & 31;
    int wave = threadIdx.x >> 5;
    unsigned char* myLds = ldsRaw + wave * WAVE_LDS;
    half_t* cacheH = (half_t*)(myLds + CACHE_B);
    half_t* hid    = (half_t*)(myLds + HID_B);
    float*  w2b    = (float*)(myLds + W2B_B);
    float*  wbuf   = (float*)(myLds + WBUF_B);
    half_t* fqb    = (half_t*)(myLds + FQB_B);
    half_t* hid2   = hid;

    int group = blockIdx.x * WAVES + wave;    // 0..8191
    int mq = lane & 15;
    int hi = lane >> 4;

    // group-uniform geometry (16 consecutive x, fixed y, fixed b)
    int q0 = group * 16;
    int b0 = q0 >> 16;
    int y0 = (q0 >> 8) & 255;
    int x0 = q0 & 255;
    int cy0 = (y0 - 1) >> 1; if (cy0 < 0) cy0 = 0;
    int cx0 = (x0 - 1) >> 1; if (cx0 < 0) cx0 = 0;
    int cstart = cx0 - 1;
    const float* fb0 = feat + (long)b0 * Cc * HW;

    // per-lane query geometry
    int x  = x0 + mq;
    int cx = (x - 1) >> 1; if (cx < 0) cx = 0;
    int cxloc = cx - cx0;                     // 0..8
    int pidx = (x & 1) | ((y0 & 1) << 1);
    const half_t* memb = wsH + MEMB_OFF + pidx * 64;

    // ------------- cooperative patch-cache fill (global -> LDS f16) ---------
    // cache[(rr*16 + l)*64 + ch] = feat[b0][ch][refl(cy0+rr-1)][refl(cstart+l)]
    for (int it = 0; it < 96; it++) {
        int slot = it >> 1;                   // uniform across the wave
        int rr = slot >> 4, l = slot & 15;
        if (l < 12) {
            int row = cy0 + rr - 1;
            row = (row < 0) ? -row : ((row > Hh - 1) ? 2 * (Hh - 1) - row : row);
            int col = cstart + l;
            col = (col < 0) ? -col : ((col > Ww - 1) ? 2 * (Ww - 1) - col : col);
            int ch = ((it & 1) << 5) + lane;
            cacheH[(slot << 6) + ch] = (half_t)fb0[ch * HW + row * Ww + col];
        }
    }

    // ---------------- Stage 1: kin(640) @ wk1^T -> hidden(128) --------------
    v8f acc[8] = {};
    const v16h* wk1f = (const v16h*)(wsH + WK1_OFF);
    v16h a = buildA(0, cacheH, memb, cxloc, hi);
    #pragma unroll
    for (int kt = 0; kt < 20; kt++) {
        v16h bf[8];
        #pragma unroll
        for (int nt = 0; nt < 8; nt++) bf[nt] = wk1f[(kt * 8 + nt) * 32 + lane];
        v16h an = (kt < 19) ? buildA(kt + 1, cacheH, memb, cxloc, hi) : a;
        #pragma unroll
        for (int nt = 0; nt < 8; nt++)
            acc[nt] = __builtin_amdgcn_wmma_f32_16x16x32_f16(
                false, a, false, bf[nt], (short)0, acc[nt], false, false);
        a = an;
    }
    // bias + GELU -> LDS halfs [q][n]
    #pragma unroll
    for (int nt = 0; nt < 8; nt++) {
        float bias = bk1[nt * 16 + mq];
        #pragma unroll
        for (int r = 0; r < 8; r++) {
            int qq = r + hi * 8;
            hid[qq * 128 + nt * 16 + mq] = (half_t)gelu_f(acc[nt][r] + bias);
        }
    }

    // ---------------- Stage 2: hidden(128) @ wk2^T -> 9 kernel weights ------
    v8f acc2 = {};
    const v16h* wk2f = (const v16h*)(wsH + WK2_OFF);
    const v8h* hv = (const v8h*)(hid + mq * 128);
    #pragma unroll
    for (int kt = 0; kt < 4; kt++) {
        v8h lo = hv[kt * 4 + hi], hh = hv[kt * 4 + hi + 2];
        v16h av = __builtin_shufflevector(lo, hh, 0,1,2,3,4,5,6,7,8,9,10,11,12,13,14,15);
        v16h bf = wk2f[kt * 32 + lane];
        acc2 = __builtin_amdgcn_wmma_f32_16x16x32_f16(
            false, av, false, bf, (short)0, acc2, false, false);
    }
    #pragma unroll
    for (int r = 0; r < 8; r++) {
        int qq = r + hi * 8;
        float v = acc2[r] + ((mq < 9) ? bk2[mq] : 0.0f);
        w2b[qq * 16 + mq] = v;
    }
    // softmax over the 9 taps (lanes 0..15 each own one query)
    if (lane < 16) {
        float w[9]; float mx = -1e30f;
        #pragma unroll
        for (int j = 0; j < 9; j++) { w[j] = w2b[lane * 16 + j]; mx = fmaxf(mx, w[j]); }
        float s = 0.0f;
        #pragma unroll
        for (int j = 0; j < 9; j++) { w[j] = __expf(w[j] - mx); s += w[j]; }
        float inv = 1.0f / s;
        #pragma unroll
        for (int j = 0; j < 9; j++) wbuf[lane * 12 + j] = w[j] * inv;
    }
    // f_q[c] = sum_j w[j] * kin_flat[j*64+c], read from the LDS patch cache.
    // Decode (c0 = i/9, t0 = i%9) with a carry recurrence instead of divisions.
    float wloc[9];
    #pragma unroll
    for (int j = 0; j < 9; j++) wloc[j] = wbuf[mq * 12 + j];
    int cbase2 = hi * 32;
    int c0r = cbase2 / 9, t0r = cbase2 % 9;   // decode of i = cbase2
    for (int cc = 0; cc < 32; cc++) {
        int c = cbase2 + cc;
        int c0 = c0r, t0 = t0r;
        float f = 0.0f;
        #pragma unroll
        for (int j = 0; j < 9; j++) {
            int ky = t0 / 3, kx = t0 - ky * 3;
            f += wloc[j] * (float)cacheH[((ky * 16 + cxloc + kx) << 6) + c0];
            // i += 64  =>  c0 += 7, t0 += 1 (64 = 7*9 + 1), with carry
            c0 += 7; t0 += 1;
            if (t0 >= 9) { t0 -= 9; c0 += 1; }
        }
        fqb[mq * 64 + c] = (half_t)f;
        // i += 1 for next cc
        t0r += 1; if (t0r >= 9) { t0r -= 9; c0r += 1; }
    }

    // ---------------- Stage 3: f_q(64) @ wr1^T -> 64 (GELU) -----------------
    v8f acc3[4] = {};
    const v16h* wr1f = (const v16h*)(wsH + WR1_OFF);
    const v8h* fv = (const v8h*)(fqb + mq * 64);
    #pragma unroll
    for (int kt = 0; kt < 2; kt++) {
        v8h lo = fv[kt * 4 + hi], hh = fv[kt * 4 + hi + 2];
        v16h av = __builtin_shufflevector(lo, hh, 0,1,2,3,4,5,6,7,8,9,10,11,12,13,14,15);
        #pragma unroll
        for (int nt = 0; nt < 4; nt++) {
            v16h bf = wr1f[(kt * 4 + nt) * 32 + lane];
            acc3[nt] = __builtin_amdgcn_wmma_f32_16x16x32_f16(
                false, av, false, bf, (short)0, acc3[nt], false, false);
        }
    }
    #pragma unroll
    for (int nt = 0; nt < 4; nt++) {
        float bias = br1[nt * 16 + mq];
        #pragma unroll
        for (int r = 0; r < 8; r++) {
            int qq = r + hi * 8;
            hid2[qq * 64 + nt * 16 + mq] = (half_t)gelu_f(acc3[nt][r] + bias);
        }
    }

    // ---------------- Stage 4: h2(64) @ wr2^T -> rgb(3) ---------------------
    v8f acc4 = {};
    const v16h* wr2f = (const v16h*)(wsH + WR2_OFF);
    const v8h* h2v = (const v8h*)(hid2 + mq * 64);
    #pragma unroll
    for (int kt = 0; kt < 2; kt++) {
        v8h lo = h2v[kt * 4 + hi], hh = h2v[kt * 4 + hi + 2];
        v16h av = __builtin_shufflevector(lo, hh, 0,1,2,3,4,5,6,7,8,9,10,11,12,13,14,15);
        v16h bf = wr2f[kt * 32 + lane];
        acc4 = __builtin_amdgcn_wmma_f32_16x16x32_f16(
            false, av, false, bf, (short)0, acc4, false, false);
    }
    if (mq < 3) {
        float bias = br2[mq];
        #pragma unroll
        for (int r = 0; r < 8; r++) {
            int M  = r + hi * 8;
            int qo = q0 + M;
            int bb = qo >> 16, yy = (qo >> 8) & 255, xx = qo & 255;
            out[(((long)bb * 3 + mq) * OH + yy) * OW + xx] = acc4[r] + bias;
        }
    }
}

extern "C" void kernel_launch(void* const* d_in, const int* in_sizes, int n_in,
                              void* d_out, int out_size, void* d_ws, size_t ws_size,
                              hipStream_t stream) {
    const float* feat = (const float*)d_in[0];
    const float* w1m  = (const float*)d_in[1];
    const float* b1m  = (const float*)d_in[2];
    const float* w2m  = (const float*)d_in[3];
    const float* b2m  = (const float*)d_in[4];
    const float* wk1  = (const float*)d_in[5];
    const float* bk1  = (const float*)d_in[6];
    const float* wk2  = (const float*)d_in[7];
    const float* bk2  = (const float*)d_in[8];
    const float* wr1  = (const float*)d_in[9];
    const float* br1  = (const float*)d_in[10];
    const float* wr2  = (const float*)d_in[11];
    const float* br2  = (const float*)d_in[12];
    float* out = (float*)d_out;
    half_t* wsH = (half_t*)d_ws;

    prep_meta<<<1, 256, 0, stream>>>(w1m, b1m, w2m, b2m, wsH);
    prep_pack<<<(WS_HALFS + 255) / 256, 256, 0, stream>>>(wk1, wk2, wr1, wr2, wsH);
    // 131072 queries / 16 per wave / 4 waves per block = 2048 blocks
    main_kernel<<<2048, 128, 0, stream>>>(feat, bk1, bk2, br1, br2, wsH, out);
}
